// Generator_17935783428276
// MI455X (gfx1250) — matrix operand order repaired and checked by reading, not compiled, not run on hardware
//
#include <hip/hip_runtime.h>
#include <hip/hip_bf16.h>

typedef __attribute__((ext_vector_type(16))) _Float16 v16h;
typedef __attribute__((ext_vector_type(8)))  float    v8f;

#define LCONV   3072      // SIZE*SIZE*3
#define NFILT   512
#define NKSTEP  (LCONV / 32)          // 96 K-steps of 32
#define NTILE_N (NFILT / 16)          // 32 column tiles
#define STEP_HALVES (NTILE_N * 512)   // halves per K-step of packed W (16384)

// ---------------------------------------------------------------------------
// Kernel 1: pre-pack conv_w (3072x512 f32) into f16 WMMA-B-fragment order.
// Tile (kt, j) covers K=[kt*32,+32), N=[j*16,+16). Within a tile, lane l
// (l%16 = N, l/16 selects K 0..7,16..23 vs 8..15,24..31) owns 16 contiguous halves:
//   P[tile*512 + l*16 + h] = W[kt*32 + 8*(l/16) + h + 8*(h/8), j*16 + l%16]
// matching the 16x16x32 f16 B-matrix VGPR layout (16 halves / lane).
// ---------------------------------------------------------------------------
__global__ __launch_bounds__(256)
void prepack_w_kernel(const float* __restrict__ W, _Float16* __restrict__ P) {
    int gid  = blockIdx.x * 256 + threadIdx.x;       // 384*256 = 3072 tiles * 32 lanes
    int lane = gid & 31;
    int tile = gid >> 5;                             // 0..3071
    int kt   = tile >> 5;                            // 0..95
    int j    = tile & 31;                            // 0..31
    int n    = j * 16 + (lane & 15);
    int kk   = kt * 32 + ((lane >> 4) << 3);
    _Float16* dst = P + (size_t)tile * 512 + lane * 16;
#pragma unroll
    for (int h = 0; h < 16; ++h)
        dst[h] = (_Float16)W[(size_t)(kk + h + ((h >> 3) << 3)) * NFILT + n];
}

// ---------------------------------------------------------------------------
// Build one 16x32 f16 A fragment of the Hankel matrix from the parity-copy of
// xp in LDS: VGPR0-3 <- xp[i..i+7], VGPR4-7 <- xp[i+16..i+23], where the uint
// offset 'uo' already encodes (i - parity)/2 into the chosen copy.
// ---------------------------------------------------------------------------
__device__ __forceinline__ v16h buildA(const uint32_t* __restrict__ src, int uo) {
    union AV { v16h v; uint32_t u[8]; } A;
#pragma unroll
    for (int q = 0; q < 4; ++q) {
        A.u[q]     = src[uo + q];
        A.u[q + 4] = src[uo + 8 + q];
    }
    return A.v;
}

// ---------------------------------------------------------------------------
// Kernel 2: fused Hankel-GEMM (WMMA f16 -> f32) + conv bias + ReLU + dense(512->4).
// WG = 32 rows x all 512 cols, 8 waves as 1(M) x 8(N): each wave owns
// 32 rows x 64 cols = 2 A-fragments x 4 B-fragments -> 8 WMMAs / K-step,
// so every B fragment feeds two WMMAs (no duplicated B loads across waves).
// B is TRIPLE-buffered (load distance 2 K-steps) so each load group has
// ~16 WMMAs of latency cover before its first consumer: enough to hide L2
// round-trips since W (3 MB/WG) streams past WGP$.
// ---------------------------------------------------------------------------
__global__ __launch_bounds__(256)
void hankel_gemm_kernel(const int*      __restrict__ example,   // (4, 3072) int32
                        const _Float16* __restrict__ P,         // packed W fragments
                        const float*    __restrict__ conv_b,    // (512,)
                        const float*    __restrict__ dense_w,   // (512,4)
                        const float*    __restrict__ dense_b,   // (4,)
                        float*          __restrict__ out)       // (12288,4)
{
    __shared__ uint32_t xp0[LCONV];        // halves xp[0..6143], pairs (2m,2m+1)
    __shared__ uint32_t xp1[LCONV];        // halves xp[1..6144], pairs (2m+1,2m+2)
    __shared__ float    cbS[NFILT];
    __shared__ float4   dwS[NFILT];
    __shared__ float    part[8][32][4];    // [wave][row][c]

    const int tid  = threadIdx.x;
    const int lane = tid & 31;
    const int wave = tid >> 5;             // 0..7 -> 64-col slice

    const int mblk = blockIdx.x;                 // 0..383
    const int b    = (mblk * 32) / LCONV;        // batch
    const int t0   = (mblk * 32) % LCONV;        // first row in batch
    const int* eb  = example + (size_t)b * LCONV;

    // ---- stage xp (with parity-shifted duplicate), conv_b, dense_w in LDS ----
    union HU { _Float16 h[2]; uint32_t u; };
    for (int m = tid; m < LCONV; m += 256) {
        int j0 = 2 * m;
        float f0 = (j0 >= LCONV)                    ? ((float)eb[j0 - LCONV] * 0.5f - 1.0f) : 0.0f;
        float f1 = (j0 + 1 >= LCONV)                ? ((float)eb[j0 + 1 - LCONV] * 0.5f - 1.0f) : 0.0f;
        float f2 = (j0 + 2 >= LCONV && j0 + 2 < 2 * LCONV)
                                                    ? ((float)eb[j0 + 2 - LCONV] * 0.5f - 1.0f) : 0.0f;
        HU a0; a0.h[0] = (_Float16)f0; a0.h[1] = (_Float16)f1; xp0[m] = a0.u;
        HU a1; a1.h[0] = (_Float16)f1; a1.h[1] = (_Float16)f2; xp1[m] = a1.u;
    }
    for (int i = tid; i < NFILT; i += 256) {
        cbS[i] = conv_b[i];
        dwS[i] = ((const float4*)dense_w)[i];
    }
    __syncthreads();

    // ---- main WMMA loop ----
    const int mrow = lane & 15;            // M within 16-row tile; also N within C tile
    const int khi  = lane >> 4;            // K-half select of the A layout
    const int pq   = lane & 1;             // parity of xp index (t0,k0 mult of 16/32)
    const uint32_t* srcP = pq ? xp1 : xp0;
    // uint offset of A-fragment row block 0 at k0=0; +8 for rows +16; +16/K-step
    const int uibase = (t0 + mrow + 8 * khi - pq) >> 1;

    v8f acc[2][4] = {};                    // 2 row-tiles x 4 col-tiles = 64 VGPRs

    const _Float16* pbase = P + (size_t)wave * 4 * 512 + (size_t)lane * 16;

#define LOADB(dst, ktv)                                                        \
    {                                                                          \
        const _Float16* pb_ = pbase + (size_t)(ktv) * STEP_HALVES;             \
        _Pragma("unroll")                                                      \
        for (int j_ = 0; j_ < 4; ++j_)                                         \
            dst[j_] = *(const v16h*)(pb_ + (size_t)j_ * 512);                  \
    }

#define COMPUTE(bf, ktv)                                                       \
    {                                                                          \
        int uo_ = uibase + (ktv) * 16;                                         \
        v16h A0_ = buildA(srcP, uo_);                                          \
        v16h A1_ = buildA(srcP, uo_ + 8);  /* rows +16 => +8 uints */          \
        _Pragma("unroll")                                                      \
        for (int j_ = 0; j_ < 4; ++j_)                                         \
            acc[0][j_] = __builtin_amdgcn_wmma_f32_16x16x32_f16(               \
                false, A0_, false, bf[j_], (short)0, acc[0][j_], false, false);\
        _Pragma("unroll")                                                      \
        for (int j_ = 0; j_ < 4; ++j_)                                         \
            acc[1][j_] = __builtin_amdgcn_wmma_f32_16x16x32_f16(               \
                false, A1_, false, bf[j_], (short)0, acc[1][j_], false, false);\
    }

    v16h bf0[4], bf1[4], bf2[4];
    LOADB(bf0, 0);
    LOADB(bf1, 1);

    for (int kt = 0; kt < NKSTEP; kt += 3) {          // 96 % 3 == 0
        // single symbolic line-prefetch far ahead (loads below are the real cover)
        int ktp = (kt + 4 < NKSTEP) ? kt + 4 : kt;
        __builtin_prefetch(pbase + (size_t)ktp * STEP_HALVES, 0, 3);

        LOADB(bf2, kt + 2);                            // kt+2 <= 95 always
        COMPUTE(bf0, kt);

        LOADB(bf0, (kt + 3 < NKSTEP) ? kt + 3 : 0);    // tail reload harmless
        COMPUTE(bf1, kt + 1);

        LOADB(bf1, (kt + 4 < NKSTEP) ? kt + 4 : 0);    // tail reload harmless
        COMPUTE(bf2, kt + 2);
    }
#undef LOADB
#undef COMPUTE

    // ---- epilogue: +conv_b, ReLU, dense(512->4) partials per lane ----
#pragma unroll
    for (int m = 0; m < 2; ++m) {
        float pr[8][4] = {};
#pragma unroll
        for (int j = 0; j < 4; ++j) {
            int n = wave * 64 + j * 16 + mrow;    // this lane's N for tile j
            float4 w  = dwS[n];
            float  cb = cbS[n];
#pragma unroll
            for (int g = 0; g < 8; ++g) {         // rows m*16 + g + 8*khi
                float v = acc[m][j][g] + cb;
                v = v > 0.0f ? v : 0.0f;
                pr[g][0] += v * w.x;
                pr[g][1] += v * w.y;
                pr[g][2] += v * w.z;
                pr[g][3] += v * w.w;
            }
        }
        // deterministic butterfly over the 16 lanes of each half-wave
#pragma unroll
        for (int g = 0; g < 8; ++g)
#pragma unroll
            for (int c = 0; c < 4; ++c) {
                float r = pr[g][c];
                r += __shfl_xor(r, 1, 32);
                r += __shfl_xor(r, 2, 32);
                r += __shfl_xor(r, 4, 32);
                r += __shfl_xor(r, 8, 32);
                pr[g][c] = r;
            }
        if (mrow == 0) {
            int rbase = m * 16 + 8 * khi;
#pragma unroll
            for (int g = 0; g < 8; ++g)
#pragma unroll
                for (int c = 0; c < 4; ++c)
                    part[wave][rbase + g][c] = pr[g][c];
        }
    }
    __syncthreads();

    if (tid < 128) {
        int row = tid >> 2;
        int c   = tid & 3;
        float v = dense_b[c];
#pragma unroll
        for (int w = 0; w < 8; ++w)
            v += part[w][row][c];
        out[(size_t)(mblk * 32 + row) * 4 + c] = v;
    }
}

// ---------------------------------------------------------------------------
extern "C" void kernel_launch(void* const* d_in, const int* in_sizes, int n_in,
                              void* d_out, int out_size, void* d_ws, size_t ws_size,
                              hipStream_t stream) {
    const int*   example = (const int*)  d_in[0];
    const float* conv_w  = (const float*)d_in[1];
    const float* conv_b  = (const float*)d_in[2];
    const float* dense_w = (const float*)d_in[3];
    const float* dense_b = (const float*)d_in[4];
    float*       out     = (float*)d_out;

    _Float16* packedW = (_Float16*)d_ws;   // needs 3 MB of workspace

    // 3072 tiles * 32 lanes = 98304 threads
    prepack_w_kernel<<<384, 256, 0, stream>>>(conv_w, packedW);
    // 12288 rows / 32 rows per WG = 384 workgroups
    hankel_gemm_kernel<<<384, 256, 0, stream>>>(example, packedW, conv_b,
                                                dense_w, dense_b, out);
}
